// GeoRwkvBlockV2_29583734735181
// MI455X (gfx1250) — matrix operand order repaired and checked by reading, not compiled, not run on hardware
//
#include <hip/hip_runtime.h>
#include <hip/hip_bf16.h>
#include <math.h>

#define Dm   256
#define Pp   32
#define Ls   1024
#define Bb   32
#define BL   (Bb*Ls)            // 32768 rows
#define BLD  ((size_t)Bb*Ls*Dm) // 8388608 elements

typedef _Float16 h8  __attribute__((ext_vector_type(8)));
typedef _Float16 h16 __attribute__((ext_vector_type(16)));
typedef float    f8  __attribute__((ext_vector_type(8)));

// ---------------------------------------------------------------- helpers
__device__ inline float block_sum_256(float v, float* sh) {
  int lane = threadIdx.x & 31, wv = threadIdx.x >> 5;
#pragma unroll
  for (int o = 16; o > 0; o >>= 1) v += __shfl_down(v, o, 32);
  if (lane == 0) sh[wv] = v;
  __syncthreads();
  if (wv == 0) {
    float t = (lane < 8) ? sh[lane] : 0.f;
#pragma unroll
    for (int o = 4; o > 0; o >>= 1) t += __shfl_down(t, o, 32);
    if (lane == 0) sh[0] = t;
  }
  __syncthreads();
  float r = sh[0];
  __syncthreads();
  return r;
}

__device__ inline float sigm(float x)  { return 1.f / (1.f + expf(-x)); }
__device__ inline float gelu_(float x) { return 0.5f * x * (1.f + erff(x * 0.70710678118654752f)); }

// -------------------------------------------------- token_shift2d + transpose
// x_grid (B,D,P,P) -> xt (B, L=P*P, D) with optional directional shift
__global__ void shift_transpose(const float* __restrict__ g, float* __restrict__ out, int do_shift) {
  __shared__ float tile[32][33];
  int b = blockIdx.z, d0 = blockIdx.y * 32, i = blockIdx.x;
  int tx = threadIdx.x & 31, ty = threadIdx.x >> 5;
#pragma unroll
  for (int r = 0; r < 4; ++r) {
    int d = d0 + ty + r * 8;
    int j = tx;
    float v;
    size_t base = ((size_t)b * Dm + d) * (Pp * Pp);
    if (!do_shift) {
      v = g[base + i * Pp + j];
    } else if (d < 52) {                       // stay
      v = g[base + i * Pp + j];
    } else if (d < 103) {                      // up: in[i-1]
      v = (i > 0) ? g[base + (i - 1) * Pp + j] : 0.f;
    } else if (d < 154) {                      // down: in[i+1]
      v = (i < Pp - 1) ? g[base + (i + 1) * Pp + j] : 0.f;
    } else if (d < 205) {                      // left: in[j-1]
      v = (j > 0) ? g[base + i * Pp + (j - 1)] : 0.f;
    } else {                                   // right: in[j+1]
      v = (j < Pp - 1) ? g[base + i * Pp + (j + 1)] : 0.f;
    }
    tile[ty + r * 8][tx] = v;
  }
  __syncthreads();
#pragma unroll
  for (int r = 0; r < 4; ++r) {
    int j = ty + r * 8;
    out[((size_t)b * Ls + i * 32 + j) * Dm + d0 + tx] = tile[tx][j];
  }
}

// -------------------------------------------- spiral gather + LayerNorm
__global__ void spiral_ln(const float* __restrict__ xt, const int* __restrict__ spiral,
                          const float* __restrict__ gg, const float* __restrict__ bb,
                          float* __restrict__ raw, float* __restrict__ lnout,
                          _Float16* __restrict__ ln16) {
  __shared__ float sh[8];
  int t = blockIdx.x, b = blockIdx.y, d = threadIdx.x;
  int row = spiral[t];
  float v = xt[((size_t)b * Ls + row) * Dm + d];
  size_t idx = ((size_t)b * Ls + t) * Dm + d;
  if (raw) raw[idx] = v;
  float mu  = block_sum_256(v, sh) * (1.f / Dm);
  float df  = v - mu;
  float var = block_sum_256(df * df, sh) * (1.f / Dm);
  float o   = df * rsqrtf(var + 1e-5f) * gg[d] + bb[d];
  lnout[idx] = o;
  ln16[idx]  = (_Float16)o;
}

// ------------------------------------------------------ alpha (routing MLP)
__global__ void alpha_kernel(const float* __restrict__ x_ln, const float* __restrict__ l_ln,
                             const float* __restrict__ gate_stats,
                             const float* __restrict__ w1, const float* __restrict__ b1,
                             const float* __restrict__ w2, const float* __restrict__ b2,
                             float* __restrict__ alpha) {
  __shared__ float pooled[2 * Dm + 3];
  __shared__ float hidden[64];
  int b = blockIdx.x, d = threadIdx.x;
  const float* xp = x_ln + (size_t)b * Ls * Dm + d;
  const float* lp = l_ln + (size_t)b * Ls * Dm + d;
  float sx = 0.f, sl = 0.f;
  for (int t = 0; t < Ls; ++t) { sx += xp[(size_t)t * Dm]; sl += lp[(size_t)t * Dm]; }
  pooled[d]      = sx * (1.f / Ls);
  pooled[Dm + d] = sl * (1.f / Ls);
  if (d < 3) pooled[2 * Dm + d] = gate_stats[b * 3 + d];
  __syncthreads();
  if (d < 64) {
    float acc = b1[d];
    for (int i = 0; i < 2 * Dm + 3; ++i) acc += pooled[i] * w1[i * 64 + d];
    hidden[d] = tanhf(acc);
  }
  __syncthreads();
  if (d == 0) {
    float acc = b2[0];
    for (int h = 0; h < 64; ++h) acc += hidden[h] * w2[h];
    alpha[b] = sigm(acc);
  }
}

__global__ void alpha_mean(const float* __restrict__ alpha, float* __restrict__ out) {
  float v = alpha[threadIdx.x];
#pragma unroll
  for (int o = 16; o > 0; o >>= 1) v += __shfl_down(v, o, 32);
  if (threadIdx.x == 0) out[0] = v * (1.f / Bb);
}

// ----------------------------------------- weight convert f32 -> f16 + T
// in (K,N) row-major f32 -> out (N,K) row-major f16
__global__ void cvt_t(const float* __restrict__ in, _Float16* __restrict__ out, int K, int N) {
  int n = K * N;
  for (int idx = blockIdx.x * blockDim.x + threadIdx.x; idx < n; idx += gridDim.x * blockDim.x) {
    int k = idx / N, c = idx - k * N;
    out[(size_t)c * K + k] = (_Float16)in[idx];
  }
}

// ------------------------------------------------------------ WMMA GEMM
// C(M,N) = epi( A(M,K)@Wt(N,K)^T [+ A2(M,K2)@Wt2(N,K2)^T] + bias )
// EPI: 0 = f32 out, 1 = f32 sigmoid out, 2 = f16 gelu(exact) out
// Block: 128 threads (4 waves); block tile 128(M)x64(N); wave tile 32x64.
__device__ inline h16 load_a16(const _Float16* __restrict__ Arow, int kb, int k0) {
  const h8* p = (const h8*)Arow;
  h8 lo = p[(kb + k0) >> 3];
  h8 hi = p[(kb + 16 + k0) >> 3];
  h16 r;
#pragma unroll
  for (int i = 0; i < 8; ++i) { r[i] = lo[i]; r[8 + i] = hi[i]; }
  return r;
}
__device__ inline h16 load_b16(const _Float16* __restrict__ Wrow, int kb, int koff) {
  const h8* p = (const h8*)Wrow;
  h8 lo = p[(kb + koff) >> 3];
  h8 hi = p[(kb + 16 + koff) >> 3];
  h16 r;
#pragma unroll
  for (int i = 0; i < 8; ++i) { r[i] = lo[i]; r[8 + i] = hi[i]; }
  return r;
}

template <int EPI>
__global__ __launch_bounds__(128) void gemm16(
    const _Float16* __restrict__ A, const _Float16* __restrict__ Wt,
    const _Float16* __restrict__ A2, const _Float16* __restrict__ Wt2,
    const float* __restrict__ bias, void* __restrict__ Cout,
    int M, int N, int K, int K2) {
  int lane = threadIdx.x & 31;
  int wave = threadIdx.x >> 5;
  int n0 = blockIdx.x * 64;
  int m0 = blockIdx.y * 128 + wave * 32;
  int k0   = (lane < 16) ? 0 : 8;   // A K-offset pattern
  int koff = (lane < 16) ? 0 : 8;   // B K-offset pattern
  f8 acc[2][4];
#pragma unroll
  for (int r = 0; r < 2; ++r)
#pragma unroll
    for (int c = 0; c < 4; ++c)
#pragma unroll
      for (int i = 0; i < 8; ++i) acc[r][c][i] = 0.f;

  int nseg = (A2 != nullptr) ? 2 : 1;
  for (int s = 0; s < nseg; ++s) {
    const _Float16* Ap = s ? A2 : A;
    const _Float16* Wp = s ? Wt2 : Wt;
    int Kc = s ? K2 : K;
    const _Float16* Arow0 = Ap + (size_t)(m0 + (lane & 15)) * Kc;
    const _Float16* Arow1 = Ap + (size_t)(m0 + 16 + (lane & 15)) * Kc;
    for (int kb = 0; kb < Kc; kb += 32) {
      // issue all loads for this K-step up front so the compiler can
      // stagger s_wait_loadcnt instead of fully serializing load->wmma
      h16 a0 = load_a16(Arow0, kb, k0);
      h16 a1 = load_a16(Arow1, kb, k0);
      h16 bt[4];
#pragma unroll
      for (int c = 0; c < 4; ++c) {
        const _Float16* Wrow = Wp + (size_t)(n0 + c * 16 + (lane & 15)) * Kc;
        bt[c] = load_b16(Wrow, kb, koff);
      }
#pragma unroll
      for (int c = 0; c < 4; ++c) {
        acc[0][c] = __builtin_amdgcn_wmma_f32_16x16x32_f16(
            false, a0, false, bt[c], (short)0, acc[0][c], false, false);
        acc[1][c] = __builtin_amdgcn_wmma_f32_16x16x32_f16(
            false, a1, false, bt[c], (short)0, acc[1][c], false, false);
      }
    }
  }
  int colb = lane & 15;
#pragma unroll
  for (int r = 0; r < 2; ++r) {
    int rbase = m0 + r * 16 + ((lane < 16) ? 0 : 8);
#pragma unroll
    for (int c = 0; c < 4; ++c) {
      int col = n0 + c * 16 + colb;
      float bi = bias ? bias[col] : 0.f;
#pragma unroll
      for (int i = 0; i < 8; ++i) {
        float v = acc[r][c][i] + bi;
        if (EPI == 1) v = sigm(v);
        if (EPI == 2) v = gelu_(v);
        size_t oi = (size_t)(rbase + i) * N + col;
        if (EPI == 2) ((_Float16*)Cout)[oi] = (_Float16)v;
        else          ((float*)Cout)[oi] = v;
      }
    }
  }
}

// -------------------------------------------- build xk/xr/xv (fwd or bwd)
__global__ void build_rkv_in(const float* __restrict__ x_ln,
                             const float* __restrict__ tmk, const float* __restrict__ tmv,
                             const float* __restrict__ tmr,
                             _Float16* __restrict__ xk16, _Float16* __restrict__ xr16,
                             _Float16* __restrict__ xv16,
                             const float* __restrict__ l_ln, _Float16* __restrict__ lrev16,
                             int bwd) {
  int t = blockIdx.x, b = blockIdx.y, d = threadIdx.x;
  size_t base = (size_t)b * Ls * Dm;
  float cur, prev;
  if (!bwd) {
    cur  = x_ln[base + (size_t)t * Dm + d];
    prev = (t > 0) ? x_ln[base + (size_t)(t - 1) * Dm + d] : 0.f;
  } else {
    cur  = x_ln[base + (size_t)(Ls - 1 - t) * Dm + d];
    prev = (t > 0) ? x_ln[base + (size_t)(Ls - t) * Dm + d] : 0.f;
    lrev16[base + (size_t)t * Dm + d] = (_Float16)l_ln[base + (size_t)(Ls - 1 - t) * Dm + d];
  }
  float mk = sigm(tmk[d]), mv = sigm(tmv[d]), mr = sigm(tmr[d]);
  size_t idx = base + (size_t)t * Dm + d;
  xk16[idx] = (_Float16)(cur * mk + prev * (1.f - mk));
  xr16[idx] = (_Float16)(cur * mr + prev * (1.f - mr));
  xv16[idx] = (_Float16)(cur * mv + prev * (1.f - mv));
}

// ---------------------------------------------------------- WKV scan
// Sequential recurrence over L=1024 steps; one thread per (b,d) channel.
// global_prefetch_b8 pulls k/v/r cachelines PF steps ahead of the serial chain.
#define PF 16
__global__ void wkv_scan(const float* __restrict__ r, const float* __restrict__ k,
                         const float* __restrict__ v, const float* __restrict__ lg,
                         const float* __restrict__ time_decay, const float* __restrict__ time_first,
                         const float* __restrict__ gsp, float* __restrict__ y, int reverse_out) {
  int b = blockIdx.x, d = threadIdx.x;
  float w  = -__expf(time_decay[d]);
  float u  = time_first[d];
  float gs = log1pf(expf(gsp[0]));  // softplus(gate_scale_param)
  float aa = 0.f, bbv = 0.f, pp = -1e38f;
  size_t base = (size_t)b * Ls * Dm + d;
  const float* rp = r + base;
  const float* kp = k + base;
  const float* vp = v + base;
  for (int t = 0; t < Ls; ++t) {
    if (t + PF < Ls) {
      __builtin_prefetch(kp + (size_t)(t + PF) * Dm, 0, 0);
      __builtin_prefetch(vp + (size_t)(t + PF) * Dm, 0, 0);
      __builtin_prefetch(rp + (size_t)(t + PF) * Dm, 0, 0);
    }
    float kk = kp[(size_t)t * Dm];
    float vv = vp[(size_t)t * Dm];
    float rr = rp[(size_t)t * Dm];
    float g  = lg[(size_t)b * Ls + t] * gs;
    float ww = u + kk;
    float p  = fmaxf(pp, ww);
    float e1 = __expf(pp - p), e2 = __expf(ww - p);
    float yv = (e1 * aa + e2 * vv) / (e1 * bbv + e2);
    int to = reverse_out ? (Ls - 1 - t) : t;
    y[(size_t)b * Ls * Dm + (size_t)to * Dm + d] = rr * yv;
    float ww2 = pp + w + g;
    float p2  = fmaxf(ww2, kk);
    float e1b = __expf(ww2 - p2), e2b = __expf(kk - p2);
    aa  = e1b * aa + e2b * vv;
    bbv = e1b * bbv + e2b;
    pp  = p2;
  }
}

// ------------------------------------------------------------ small fused
__global__ void ysum_kernel(const float* __restrict__ yf, const float* __restrict__ yb,
                            _Float16* __restrict__ ysum16) {
  for (size_t i = (size_t)blockIdx.x * blockDim.x + threadIdx.x; i < BLD;
       i += (size_t)gridDim.x * blockDim.x)
    ysum16[i] = (_Float16)(yf[i] + yb[i]);
}

// x2 = x + a*out_fast + (1-a)*out_fusion (in place on x), then xc = LN(x2)
__global__ void mix_ln(float* __restrict__ x, const float* __restrict__ of,
                       const float* __restrict__ ofu, const float* __restrict__ alpha,
                       const float* __restrict__ gg, const float* __restrict__ be,
                       float* __restrict__ xc) {
  __shared__ float sh[8];
  int t = blockIdx.x, b = blockIdx.y, d = threadIdx.x;
  size_t idx = ((size_t)b * Ls + t) * Dm + d;
  float a = alpha[b];
  float v = x[idx] + a * of[idx] + (1.f - a) * ofu[idx];
  x[idx] = v;
  float mu  = block_sum_256(v, sh) * (1.f / Dm);
  float df  = v - mu;
  float var = block_sum_256(df * df, sh) * (1.f / Dm);
  xc[idx] = df * rsqrtf(var + 1e-5f) * gg[d] + be[d];
}

__global__ void build_cm_in(const float* __restrict__ xc, const float* __restrict__ cm_mk,
                            const float* __restrict__ cm_mr,
                            _Float16* __restrict__ xk16, _Float16* __restrict__ xr16) {
  int t = blockIdx.x, b = blockIdx.y, d = threadIdx.x;
  size_t base = (size_t)b * Ls * Dm;
  float cur  = xc[base + (size_t)t * Dm + d];
  float prev = (t > 0) ? xc[base + (size_t)(t - 1) * Dm + d] : 0.f;
  float mk = fminf(fmaxf(cm_mk[d], 0.f), 1.f);
  float mr = fminf(fmaxf(cm_mr[d], 0.f), 1.f);
  size_t idx = base + (size_t)t * Dm + d;
  xk16[idx] = (_Float16)(cur * mk + prev * (1.f - mk));
  xr16[idx] = (_Float16)(cur * mr + prev * (1.f - mr));
}

// out grid (B,D,P,P) = (x2 + rc*vc) scattered through inverse spiral
__global__ void final_out(const float* __restrict__ x2, const float* __restrict__ rc,
                          const float* __restrict__ vc, const int* __restrict__ inv_spiral,
                          float* __restrict__ out) {
  int idx = blockIdx.x * blockDim.x + threadIdx.x;  // < 2^23
  int j = idx & 31, i = (idx >> 5) & 31, d = (idx >> 10) & 255, b = idx >> 18;
  int t = inv_spiral[(i << 5) | j];
  size_t o = ((size_t)b * Ls + t) * Dm + d;
  out[idx] = x2[o] + rc[o] * vc[o];
}

// ---------------------------------------------------------------- launcher
static void launch_gemm(int epi, const _Float16* A, const _Float16* Wt,
                        const _Float16* A2, const _Float16* Wt2, const float* bias,
                        void* C, int M, int N, int K, int K2, hipStream_t s) {
  dim3 g(N / 64, M / 128), b(128);
  switch (epi) {
    case 0: gemm16<0><<<g, b, 0, s>>>(A, Wt, A2, Wt2, bias, C, M, N, K, K2); break;
    case 1: gemm16<1><<<g, b, 0, s>>>(A, Wt, A2, Wt2, bias, C, M, N, K, K2); break;
    case 2: gemm16<2><<<g, b, 0, s>>>(A, Wt, A2, Wt2, bias, C, M, N, K, K2); break;
  }
}

enum {
  I_LN_X_G = 0, I_LN_X_B, I_LN_L_G, I_LN_L_B,
  I_FMLP_W1, I_FMLP_B1, I_FMLP_W2, I_FMLP_B2,
  I_TMK, I_TMV, I_TMR, I_TIME_DECAY, I_TIME_FIRST,
  I_W_RX, I_W_KX, I_W_VX, I_W_RL, I_W_KL, I_W_OUT,
  I_LN_C_G, I_LN_C_B, I_CM_MK, I_CM_MR, I_CM_KEY, I_CM_VAL, I_CM_REC,
  I_RT_W1, I_RT_B1, I_RT_W2, I_RT_B2, I_GATE_SCALE,
  I_X_GRID, I_L_GRID, I_LG_F, I_LG_B, I_GATE_STATS, I_SPIRAL, I_INV_SPIRAL
};

extern "C" void kernel_launch(void* const* d_in, const int* in_sizes, int n_in,
                              void* d_out, int out_size, void* d_ws, size_t ws_size,
                              hipStream_t stream) {
  (void)in_sizes; (void)n_in; (void)out_size; (void)ws_size;
  char* wsb = (char*)d_ws;
  size_t off = 0;
  auto take = [&](size_t bytes) -> char* {
    char* p = wsb + off;
    off += (bytes + 255) & ~(size_t)255;
    return p;
  };
  const size_t F32 = BLD * 4, F16 = BLD * 2;
  float* S0 = (float*)take(F32);          // xt  -> y_fwd
  float* S1 = (float*)take(F32);          // lt  -> y_bwd
  float* S2 = (float*)take(F32);          // x   -> x2 (in place)
  float* S3 = (float*)take(F32);          // x_ln -> xc
  float* S4 = (float*)take(F32);          // l_ln -> out_fusion -> rc
  float* S5 = (float*)take(F32);          // out_fast -> vc
  float* S7 = (float*)take(F32);          // r (fwd then bwd)
  float* S8 = (float*)take(F32);          // k
  float* S9 = (float*)take(F32);          // v
  _Float16* S10 = (_Float16*)take((size_t)BL * 768 * 2);  // h1_16 -> {ysum16,cmxk16,cmxr16}
  _Float16* S11 = (_Float16*)take((size_t)BL * 1024 * 2); // kc16
  _Float16* H0 = (_Float16*)take(F16);    // x_ln16
  _Float16* H1 = (_Float16*)take(F16);    // l_ln16
  _Float16* H2 = (_Float16*)take(F16);    // l_rev16
  _Float16* H3 = (_Float16*)take(F16);    // xk16
  _Float16* H4 = (_Float16*)take(F16);    // xr16
  _Float16* H5 = (_Float16*)take(F16);    // xv16
  _Float16* w1t   = (_Float16*)take((size_t)256 * 768 * 2);
  _Float16* w2t   = (_Float16*)take((size_t)768 * 256 * 2);
  _Float16* wrxt  = (_Float16*)take((size_t)256 * 256 * 2);
  _Float16* wkxt  = (_Float16*)take((size_t)256 * 256 * 2);
  _Float16* wvxt  = (_Float16*)take((size_t)256 * 256 * 2);
  _Float16* wrlt  = (_Float16*)take((size_t)256 * 256 * 2);
  _Float16* wklt  = (_Float16*)take((size_t)256 * 256 * 2);
  _Float16* woutt = (_Float16*)take((size_t)256 * 256 * 2);
  _Float16* ckeyt = (_Float16*)take((size_t)256 * 1024 * 2);
  _Float16* cvalt = (_Float16*)take((size_t)1024 * 256 * 2);
  _Float16* crect = (_Float16*)take((size_t)256 * 256 * 2);
  float* alphav = (float*)take(128);

  _Float16* ysum16 = S10;                 // 16MB
  _Float16* cmxk16 = S10 + BLD;           // next 16MB
  _Float16* cmxr16 = S10 + 2 * BLD;       // next 16MB

  auto in = [&](int i) { return (const float*)d_in[i]; };
  const int* spiral     = (const int*)d_in[I_SPIRAL];
  const int* inv_spiral = (const int*)d_in[I_INV_SPIRAL];
  float* out = (float*)d_out;

  // --- weights f32 -> f16 transposed (N,K) ---
  auto cvt = [&](const float* src, _Float16* dst, int K, int N) {
    int n = K * N;
    cvt_t<<<(n + 255) / 256, 256, 0, stream>>>(src, dst, K, N);
  };
  cvt(in(I_FMLP_W1), w1t, 256, 768);
  cvt(in(I_FMLP_W2), w2t, 768, 256);
  cvt(in(I_W_RX), wrxt, 256, 256);
  cvt(in(I_W_KX), wkxt, 256, 256);
  cvt(in(I_W_VX), wvxt, 256, 256);
  cvt(in(I_W_RL), wrlt, 256, 256);
  cvt(in(I_W_KL), wklt, 256, 256);
  cvt(in(I_W_OUT), woutt, 256, 256);
  cvt(in(I_CM_KEY), ckeyt, 256, 1024);
  cvt(in(I_CM_VAL), cvalt, 1024, 256);
  cvt(in(I_CM_REC), crect, 256, 256);

  // --- token_shift2d + transpose; plain transpose for l ---
  dim3 stg(Pp, Dm / 32, Bb);
  shift_transpose<<<stg, 256, 0, stream>>>(in(I_X_GRID), S0, 1);
  shift_transpose<<<stg, 256, 0, stream>>>(in(I_L_GRID), S1, 0);

  // --- spiral gather + LayerNorm ---
  dim3 glb(Ls, Bb);
  spiral_ln<<<glb, 256, 0, stream>>>(S0, spiral, in(I_LN_X_G), in(I_LN_X_B), S2, S3, H0);
  spiral_ln<<<glb, 256, 0, stream>>>(S1, spiral, in(I_LN_L_G), in(I_LN_L_B), nullptr, S4, H1);

  // --- routing alpha ---
  alpha_kernel<<<Bb, 256, 0, stream>>>(S3, S4, in(I_GATE_STATS), in(I_RT_W1), in(I_RT_B1),
                                       in(I_RT_W2), in(I_RT_B2), alphav);
  alpha_mean<<<1, 32, 0, stream>>>(alphav, out + BLD);

  // --- fast MLP: h1 = gelu(x_ln@w1+b1) (f16), out_fast = h1@w2+b2 (f32) ---
  launch_gemm(2, H0, w1t, nullptr, nullptr, in(I_FMLP_B1), S10, BL, 768, 256, 0, stream);
  launch_gemm(0, S10, w2t, nullptr, nullptr, in(I_FMLP_B2), S5, BL, 256, 768, 0, stream);

  // --- forward r,k,v ---
  build_rkv_in<<<glb, 256, 0, stream>>>(S3, in(I_TMK), in(I_TMV), in(I_TMR), H3, H4, H5,
                                        S4, nullptr, 0);
  launch_gemm(1, H4, wrxt, H1, wrlt, nullptr, S7, BL, 256, 256, 256, stream);  // r=sigmoid
  launch_gemm(0, H3, wkxt, H1, wklt, nullptr, S8, BL, 256, 256, 256, stream);  // k
  launch_gemm(0, H5, wvxt, nullptr, nullptr, nullptr, S9, BL, 256, 256, 0, stream);  // v
  wkv_scan<<<Bb, 256, 0, stream>>>(S7, S8, S9, in(I_LG_F), in(I_TIME_DECAY), in(I_TIME_FIRST),
                                   in(I_GATE_SCALE), S0, 0);

  // --- backward r,k,v (reversed sequence) ---
  build_rkv_in<<<glb, 256, 0, stream>>>(S3, in(I_TMK), in(I_TMV), in(I_TMR), H3, H4, H5,
                                        S4, H2, 1);
  launch_gemm(1, H4, wrxt, H2, wrlt, nullptr, S7, BL, 256, 256, 256, stream);
  launch_gemm(0, H3, wkxt, H2, wklt, nullptr, S8, BL, 256, 256, 256, stream);
  launch_gemm(0, H5, wvxt, nullptr, nullptr, nullptr, S9, BL, 256, 256, 0, stream);
  wkv_scan<<<Bb, 256, 0, stream>>>(S7, S8, S9, in(I_LG_B), in(I_TIME_DECAY), in(I_TIME_FIRST),
                                   in(I_GATE_SCALE), S1, 1);

  // --- fusion: (y_fwd + y_bwd) @ w_out ---
  ysum_kernel<<<4096, 256, 0, stream>>>(S0, S1, ysum16);
  launch_gemm(0, ysum16, woutt, nullptr, nullptr, nullptr, S4, BL, 256, 256, 0, stream);

  // --- mix + channel-mix LN ---
  mix_ln<<<glb, 256, 0, stream>>>(S2, S5, S4, alphav, in(I_LN_C_G), in(I_LN_C_B), S3);
  build_cm_in<<<glb, 256, 0, stream>>>(S3, in(I_CM_MK), in(I_CM_MR), cmxk16, cmxr16);
  launch_gemm(2, cmxk16, ckeyt, nullptr, nullptr, nullptr, S11, BL, 1024, 256, 0, stream);  // kc=gelu
  launch_gemm(0, S11, cvalt, nullptr, nullptr, nullptr, S5, BL, 256, 1024, 0, stream);      // vc
  launch_gemm(1, cmxr16, crect, nullptr, nullptr, nullptr, S4, BL, 256, 256, 0, stream);    // rc

  // --- final scatter to (B,D,P,P) ---
  final_out<<<(Bb * Dm * Pp * Pp) / 256, 256, 0, stream>>>(S2, S4, S5, inv_spiral, out);
}
